// RelativeAttention_38500086841684
// MI455X (gfx1250) — matrix operand order repaired and checked by reading, not compiled, not run on hardware
//
#include <hip/hip_runtime.h>
#include <hip/hip_bf16.h>
#include <stdint.h>

// ---------------- CDNA5 WMMA types ----------------
typedef __attribute__((ext_vector_type(16))) __bf16          bf16x16;
typedef __attribute__((ext_vector_type(8)))  float           floatx8;
typedef __attribute__((ext_vector_type(8)))  unsigned short  ushort8;

union FragAB {
  bf16x16 v;
  ushort8 h[2];
};

__device__ __forceinline__ unsigned short f2bf(float f) {
  union { float f; unsigned u; } a; a.f = f;
  unsigned r = a.u + 0x7FFFu + ((a.u >> 16) & 1u);   // round-to-nearest-even
  return (unsigned short)(r >> 16);
}
__device__ __forceinline__ float bf2f(unsigned short b) {
  union { unsigned u; float f; } a; a.u = ((unsigned)b) << 16; return a.f;
}

// Problem constants
#define NS   4
#define LQ   1024
#define SS   512
#define HH   8
#define DD   64
#define CC   2
#define DCX  32
#define PP   128
#define NE   (NS * LQ * SS)          // 2097152 elements of content tensors

// ---------------- conversion kernels ----------------
__global__ __launch_bounds__(256) void k_f32_to_bf16(
    const float* __restrict__ src, unsigned short* __restrict__ dst, int n) {
  int i = blockIdx.x * blockDim.x + threadIdx.x;
  if (i < n) dst[i] = f2bf(src[i]);
}

// W [512 in, 512 out] f32 -> Wt [512 out, 512 in] bf16 (B-fragment friendly)
__global__ __launch_bounds__(256) void k_transpose_w512(
    const float* __restrict__ W, unsigned short* __restrict__ Wt) {
  int t = blockIdx.x * blockDim.x + threadIdx.x;  // 262144
  int i = t >> 9, o = t & 511;
  Wt[o * 512 + i] = f2bf(W[t]);
}

// ---------------- positional tables pk/pq [P,C,H,DC] f32 ----------------
__global__ __launch_bounds__(256) void k_pkpq(
    const float* __restrict__ pos_emb,
    const float* __restrict__ Wpk, const float* __restrict__ bpk,
    const float* __restrict__ Wpq, const float* __restrict__ bpq,
    float* __restrict__ pk, float* __restrict__ pq) {
  int t = blockIdx.x * blockDim.x + threadIdx.x;  // 65536 = P*512
  int p  = t >> 9;
  int o  = t & 511;       // c*256 + o2
  int c  = o >> 8;
  int o2 = o & 255;       // h*32 + dc
  const float* x  = pos_emb + p * 128 + c * 64;
  const float* wk = Wpk + c * 64 * 256 + o2;
  const float* wq = Wpq + c * 64 * 256 + o2;
  float sk = bpk[c * 256 + o2];
  float sq = bpq[c * 256 + o2];
  #pragma unroll 8
  for (int i = 0; i < 64; ++i) {
    float xv = x[i];
    sk += xv * wk[i * 256];
    sq += xv * wq[i * 256];
  }
  pk[t] = sk; pq[t] = sq;
}

// ---------------- projection GEMM: [4096,512] @ [512,512] + bias ----------------
// transposed_v==0: out bf16 row-major [4096,512]
// transposed_v==1: out bf16 cvT layout [N,H,D,J] = (n*512+col)*1024 + j
__global__ __launch_bounds__(256) void k_proj_gemm(
    const unsigned short* __restrict__ X, const unsigned short* __restrict__ Wt,
    const float* __restrict__ bias, unsigned short* __restrict__ out,
    int transposed_v) {
  int wave = blockIdx.x * (blockDim.x >> 5) + (threadIdx.x >> 5); // 8192 waves
  int lane = threadIdx.x & 31;
  int m = lane & 15, g = lane >> 4;
  int mt = wave >> 5;   // 256 row tiles of 16
  int nt = wave & 31;   // 32 col tiles of 16
  const unsigned short* arow = X  + (mt * 16 + m) * 512;
  const unsigned short* bcol = Wt + (nt * 16 + m) * 512;
  floatx8 acc = {0.f,0.f,0.f,0.f,0.f,0.f,0.f,0.f};
  for (int k = 0; k < 512; k += 32) {
    FragAB a, b;
    a.h[0] = *(const ushort8*)(arow + k + g * 8);       // K 0-7 / 8-15
    a.h[1] = *(const ushort8*)(arow + k + 16 + g * 8);  // K 16-23 / 24-31
    b.h[0] = *(const ushort8*)(bcol + k + g * 8);       // K 0-7 / 8-15
    b.h[1] = *(const ushort8*)(bcol + k + 16 + g * 8);  // K 16-23 / 24-31
    acc = __builtin_amdgcn_wmma_f32_16x16x32_bf16(false, a.v, false, b.v,
                                                  (short)0, acc, false, false);
  }
  int col = nt * 16 + m;
  float bv = bias[col];
  #pragma unroll
  for (int r = 0; r < 8; ++r) {
    int row = mt * 16 + r + g * 8;
    float vv = acc[r] + bv;
    if (!transposed_v) {
      out[row * 512 + col] = f2bf(vv);
    } else {
      out[((row >> 10) * 512 + col) * 1024 + (row & 1023)] = f2bf(vv);
    }
  }
}

// ---------------- c2p_all [N,I,P,C,H]: dot(cq_split, pk) ----------------
__global__ __launch_bounds__(256) void k_c2p_all(
    const unsigned short* __restrict__ cq, const float* __restrict__ pk,
    float* __restrict__ c2p_all) {
  __shared__ float q[512];
  int ni = blockIdx.x;                       // n*1024 + i
  for (int t = threadIdx.x; t < 512; t += blockDim.x)
    q[t] = bf2f(cq[ni * 512 + t]);
  __syncthreads();
  for (int t = threadIdx.x; t < 2048; t += blockDim.x) {  // p*16 + c*8 + h
    int p = t >> 4, c = (t >> 3) & 1, h = t & 7;
    const float* pkr = pk + p * 512 + c * 256 + h * 32;
    const float* qr  = q + h * 64 + c * 32;
    float s = 0.f;
    #pragma unroll
    for (int d = 0; d < 32; ++d) s += qr[d] * pkr[d];
    c2p_all[ni * 2048 + t] = s;
  }
}

// ---------------- p2c_all [N,P,J,C,H]: dot(pq, ck_split) ----------------
__global__ __launch_bounds__(256) void k_p2c_all(
    const unsigned short* __restrict__ ck, const float* __restrict__ pq,
    float* __restrict__ p2c_all) {
  __shared__ float kr[512];
  int nj = blockIdx.x;
  int n = nj >> 10, j = nj & 1023;
  for (int t = threadIdx.x; t < 512; t += blockDim.x)
    kr[t] = bf2f(ck[nj * 512 + t]);
  __syncthreads();
  for (int t = threadIdx.x; t < 2048; t += blockDim.x) {
    int p = t >> 4, c = (t >> 3) & 1, h = t & 7;
    const float* pqr = pq + p * 512 + c * 256 + h * 32;
    const float* kk  = kr + h * 64 + c * 32;
    float s = 0.f;
    #pragma unroll
    for (int d = 0; d < 32; ++d) s += kk[d] * pqr[d];
    p2c_all[((n * 128 + p) * 1024 + j) * 16 + (t & 15)] = s;
  }
}

// ---------------- scores: WMMA QK^T + gathered c2p/p2c + mask ----------------
__global__ __launch_bounds__(256) void k_score(
    const unsigned short* __restrict__ cq, const unsigned short* __restrict__ ck,
    const float* __restrict__ c2p_all, const float* __restrict__ p2c_all,
    const int* __restrict__ rp, const unsigned char* __restrict__ mask,
    const float* __restrict__ head_bias, float* __restrict__ attn) {
  int b = blockIdx.x;                          // N*64*64 = 16384
  int n = b >> 12, it = (b >> 6) & 63, jt = b & 63;
  int h = threadIdx.x >> 5;                    // one head per wave
  int lane = threadIdx.x & 31, m = lane & 15, g = lane >> 4;
  const unsigned short* arow = cq + ((n * 1024 + it * 16 + m) * 512 + h * 64);
  const unsigned short* bcol = ck + ((n * 1024 + jt * 16 + m) * 512 + h * 64);
  floatx8 acc = {0.f,0.f,0.f,0.f,0.f,0.f,0.f,0.f};
  #pragma unroll
  for (int k = 0; k < 64; k += 32) {
    FragAB a, bfr;
    a.h[0]   = *(const ushort8*)(arow + k + g * 8);
    a.h[1]   = *(const ushort8*)(arow + k + 16 + g * 8);
    bfr.h[0] = *(const ushort8*)(bcol + k + g * 8);
    bfr.h[1] = *(const ushort8*)(bcol + k + 16 + g * 8);
    acc = __builtin_amdgcn_wmma_f32_16x16x32_bf16(false, a.v, false, bfr.v,
                                                  (short)0, acc, false, false);
  }
  int j = jt * 16 + m;
  float hb = head_bias[h];
  bool mk = mask[n * 1024 + j] != 0;
  #pragma unroll
  for (int r = 0; r < 8; ++r) {
    int i = it * 16 + r + g * 8;
    const int* rpp = rp + (((n * 1024 + i) * 1024 + j) << 1);
    int p0 = rpp[0]; p0 = p0 < 0 ? 0 : (p0 > 127 ? 127 : p0);
    int p1 = rpp[1]; p1 = p1 < 0 ? 0 : (p1 > 127 ? 127 : p1);
    float v = acc[r];
    int cbase = (n * 1024 + i) * 2048;
    v += c2p_all[cbase + p0 * 16 + h] + c2p_all[cbase + p1 * 16 + 8 + h];
    v += p2c_all[((n * 128 + p0) * 1024 + j) * 16 + h]
       + p2c_all[((n * 128 + p1) * 1024 + j) * 16 + 8 + h];
    v = v * 0.125f + hb;                       // scale = 1/sqrt(64)
    v = mk ? v : -10000.0f;
    attn[((n * 1024 + i) * 1024 + j) * 8 + h] = v;
  }
}

// ---------------- softmax over j -> bf16 weights [N,H,I,J] ----------------
__global__ __launch_bounds__(256) void k_softmax(
    const float* __restrict__ attn, unsigned short* __restrict__ wts) {
  int b = blockIdx.x;                          // n*1024 + i
  int n = b >> 10, i = b & 1023;
  int h = threadIdx.x >> 5, lane = threadIdx.x & 31;
  const float* row = attn + (size_t)b * 8192 + h;   // stride 8 over j
  float mx = -3.0e38f;
  for (int j = lane; j < 1024; j += 32) mx = fmaxf(mx, row[j * 8]);
  #pragma unroll
  for (int off = 16; off > 0; off >>= 1) mx = fmaxf(mx, __shfl_xor(mx, off, 32));
  float sum = 0.f;
  for (int j = lane; j < 1024; j += 32) sum += __expf(row[j * 8] - mx);
  #pragma unroll
  for (int off = 16; off > 0; off >>= 1) sum += __shfl_xor(sum, off, 32);
  float inv = 1.0f / sum;
  unsigned short* w = wts + ((size_t)(n * 8 + h) * 1024 + i) * 1024;
  for (int j = lane; j < 1024; j += 32) w[j] = f2bf(__expf(row[j * 8] - mx) * inv);
}

// ---------------- output GEMM: weights[N,H,I,J] @ cvT[N,H,D,J]^T ----------------
__global__ __launch_bounds__(256) void k_out_gemm(
    const unsigned short* __restrict__ wts, const unsigned short* __restrict__ cvT,
    float* __restrict__ out) {
  int wave = blockIdx.x * (blockDim.x >> 5) + (threadIdx.x >> 5); // 8192 waves
  int lane = threadIdx.x & 31, m = lane & 15, g = lane >> 4;
  int nh = wave >> 8;                       // 32 (n,h) pairs
  int tile = wave & 255;
  int mt = tile >> 2, nt = tile & 3;        // 64 i-tiles x 4 d-tiles
  const unsigned short* arow = wts + ((size_t)(nh * 1024) + mt * 16 + m) * 1024;
  const unsigned short* bcol = cvT + ((size_t)(nh * 64) + nt * 16 + m) * 1024;
  floatx8 acc = {0.f,0.f,0.f,0.f,0.f,0.f,0.f,0.f};
  for (int k = 0; k < 1024; k += 32) {
    FragAB a, bfr;
    a.h[0]   = *(const ushort8*)(arow + k + g * 8);
    a.h[1]   = *(const ushort8*)(arow + k + 16 + g * 8);
    bfr.h[0] = *(const ushort8*)(bcol + k + g * 8);
    bfr.h[1] = *(const ushort8*)(bcol + k + 16 + g * 8);
    acc = __builtin_amdgcn_wmma_f32_16x16x32_bf16(false, a.v, false, bfr.v,
                                                  (short)0, acc, false, false);
  }
  int n = nh >> 3, h = nh & 7;
  int col = h * 64 + nt * 16 + m;
  #pragma unroll
  for (int r = 0; r < 8; ++r) {
    int i = mt * 16 + r + g * 8;
    out[(n * 1024 + i) * 512 + col] = acc[r];
  }
}

// ---------------- host orchestration ----------------
extern "C" void kernel_launch(void* const* d_in, const int* in_sizes, int n_in,
                              void* d_out, int out_size, void* d_ws, size_t ws_size,
                              hipStream_t stream) {
  (void)in_sizes; (void)n_in; (void)out_size; (void)ws_size;

  const float* Xq  = (const float*)d_in[0];
  const float* Xk  = (const float*)d_in[1];
  const float* Xv  = (const float*)d_in[2];
  const unsigned char* mask = (const unsigned char*)d_in[3];
  const int*   rp  = (const int*)d_in[4];
  const float* pos_emb = (const float*)d_in[5];
  const float* Wq = (const float*)d_in[6];   const float* bq = (const float*)d_in[7];
  const float* Wk = (const float*)d_in[8];   const float* bk = (const float*)d_in[9];
  const float* Wv = (const float*)d_in[10];  const float* bv = (const float*)d_in[11];
  const float* Wpk = (const float*)d_in[12]; const float* bpk = (const float*)d_in[13];
  const float* Wpq = (const float*)d_in[14]; const float* bpq = (const float*)d_in[15];
  const float* head_bias = (const float*)d_in[16];

  float* out_main = (float*)d_out;                       // [4,1024,512]
  float* out_attn = out_main + (size_t)NS * LQ * SS;     // [4,1024,1024,8]

  // workspace carve (256B aligned): ~154 MB total
  char* wsp = (char*)d_ws;
  auto carve = [&](size_t bytes) -> void* {
    void* p = (void*)wsp; wsp += (bytes + 255) & ~(size_t)255; return p;
  };
  unsigned short* Xq_bf = (unsigned short*)carve((size_t)NE * 2);
  unsigned short* Xk_bf = (unsigned short*)carve((size_t)NE * 2);
  unsigned short* Xv_bf = (unsigned short*)carve((size_t)NE * 2);
  unsigned short* Wtq   = (unsigned short*)carve((size_t)512 * 512 * 2);
  unsigned short* Wtk   = (unsigned short*)carve((size_t)512 * 512 * 2);
  unsigned short* Wtv   = (unsigned short*)carve((size_t)512 * 512 * 2);
  unsigned short* cq_bf = (unsigned short*)carve((size_t)NE * 2);
  unsigned short* ck_bf = (unsigned short*)carve((size_t)NE * 2);
  unsigned short* cvT   = (unsigned short*)carve((size_t)NE * 2);
  float*          pk    = (float*)carve((size_t)PP * 512 * 4);
  float*          pq    = (float*)carve((size_t)PP * 512 * 4);
  float*          c2p   = (float*)carve((size_t)NS * LQ * PP * 16 * 4);   // 32 MB
  float*          p2c   = (float*)carve((size_t)NS * PP * LQ * 16 * 4);   // 32 MB
  unsigned short* wts   = (unsigned short*)carve((size_t)NS * HH * LQ * LQ * 2); // 64 MB

  const dim3 B256(256);

  // 1) conversions
  k_f32_to_bf16<<<NE / 256, B256, 0, stream>>>(Xq, Xq_bf, NE);
  k_f32_to_bf16<<<NE / 256, B256, 0, stream>>>(Xk, Xk_bf, NE);
  k_f32_to_bf16<<<NE / 256, B256, 0, stream>>>(Xv, Xv_bf, NE);
  k_transpose_w512<<<(512 * 512) / 256, B256, 0, stream>>>(Wq, Wtq);
  k_transpose_w512<<<(512 * 512) / 256, B256, 0, stream>>>(Wk, Wtk);
  k_transpose_w512<<<(512 * 512) / 256, B256, 0, stream>>>(Wv, Wtv);

  // 2) positional tables
  k_pkpq<<<(PP * 512) / 256, B256, 0, stream>>>(pos_emb, Wpk, bpk, Wpq, bpq, pk, pq);

  // 3) WMMA projections (8192 tile-waves each, 8 waves/block)
  k_proj_gemm<<<1024, B256, 0, stream>>>(Xq_bf, Wtq, bq, cq_bf, 0);
  k_proj_gemm<<<1024, B256, 0, stream>>>(Xk_bf, Wtk, bk, ck_bf, 0);
  k_proj_gemm<<<1024, B256, 0, stream>>>(Xv_bf, Wtv, bv, cvT, 1);

  // 4) positional score tables
  k_c2p_all<<<NS * LQ, B256, 0, stream>>>(cq_bf, pk, c2p);
  k_p2c_all<<<NS * LQ, B256, 0, stream>>>(ck_bf, pq, p2c);

  // 5) masked scores (second output)
  k_score<<<NS * 64 * 64, B256, 0, stream>>>(cq_bf, ck_bf, c2p, p2c, rp, mask,
                                             head_bias, out_attn);

  // 6) softmax -> bf16 weights
  k_softmax<<<NS * LQ, B256, 0, stream>>>(out_attn, wts);

  // 7) WMMA output GEMM (first output)
  k_out_gemm<<<1024, B256, 0, stream>>>(wts, cvT, out_main);
}
